// RPN_53249004535849
// MI455X (gfx1250) — matrix operand order repaired and checked by reading, not compiled, not run on hardware
//
#include <hip/hip_runtime.h>
#include <hip/hip_bf16.h>

typedef __attribute__((ext_vector_type(16))) _Float16 v16h;
typedef __attribute__((ext_vector_type(8)))  _Float16 v8h;
typedef __attribute__((ext_vector_type(8)))  float    v8f;

#define HH 96
#define WW 96
#define CC 512
#define PW 98                  // padded width/height
#define KTOT 4608              // 3*3*512
#define NPOS 9216              // 96*96
#define NANCH 82944            // 9216*9
#define MAXPRE 12000
#define MAXPOST 2000
#define NEGF (-1e9f)
#define IOUT 0.7f
#define MINSZ 16.0f
#define IMSZ 1536.0f

// ---------------- workspace layout (byte offsets, all 64B-aligned) ----------
#define O_FEATPAD 0u                         // f16 98*98*512      = 9,834,496 B
#define O_WT      9834496u                   // f16 512*4608       = 4,718,592 B
#define O_Y       14553088u                  // f16 9216*512       = 9,437,184 B
#define O_CLSLOG  23990272u                  // f32 9216*9         =   331,776 B
#define O_BOXES   24322048u                  // f32 82944*4        = 1,327,104 B
#define O_OBJ     25649152u                  // f32 82944          =   331,776 B
#define O_VALID   25980928u                  // f32 82944          =   331,776 B
#define O_SELS    26312704u                  // f32 12000          =    48,000 B
#define O_SELB    26360704u                  // f32 12000*4        =   192,000 B
#define O_SELA    26552704u                  // f32 12000          =    48,000 B
#define O_HIST    26600704u                  // u32 65536          =   262,144 B
#define O_META    26862848u                  // i32 8

// ---------------------------------------------------------------------------
// K1: zero-pad feature map to f16 and transpose conv weights to [cout][k] f16
// ---------------------------------------------------------------------------
__global__ __launch_bounds__(256)
void prep_kernel(const float* __restrict__ feat, const float* __restrict__ convW,
                 _Float16* __restrict__ featPad, _Float16* __restrict__ wT) {
    size_t i = (size_t)blockIdx.x * 256 + threadIdx.x;
    const size_t NP = (size_t)PW * PW * CC;
    const size_t NW = (size_t)CC * KTOT;
    if (i < NP) {
        size_t c  = i & 511;
        size_t px = i >> 9;
        int x = (int)(px % PW), y = (int)(px / PW);
        _Float16 v = (_Float16)0.0f;
        if (y >= 1 && y <= HH && x >= 1 && x <= WW)
            v = (_Float16)feat[(((size_t)(y - 1)) * WW + (x - 1)) * CC + c];
        featPad[i] = v;
    } else if (i < NP + NW) {
        size_t j = i - NP;                 // j = n*4608 + k
        size_t n = j / KTOT;
        size_t k = j - n * KTOT;           // k = tap*512 + cin
        wT[j] = (_Float16)convW[k * CC + n];   // HWIO: [(tap*512+cin)*512 + cout]
    }
}

// ---------------------------------------------------------------------------
// K2: implicit-GEMM 3x3 conv via v_wmma_f32_16x16x32_f16, bias+ReLU, y -> f16
//     block = 256 thr = 8 waves; wave tile 16(M)x64(N); block tile 128x64
// ---------------------------------------------------------------------------
__global__ __launch_bounds__(256)
void gemm_conv_kernel(const _Float16* __restrict__ featPad,
                      const _Float16* __restrict__ wT,
                      const float* __restrict__ convB,
                      _Float16* __restrict__ yOut) {
    __shared__ _Float16 bTile[64 * 512];          // 64 KB: one tap's B slab

    const int tid    = threadIdx.x;
    const int wave   = tid >> 5;
    const int lane   = tid & 31;
    const int laneLo = lane & 15;
    const int m0 = blockIdx.x * 128 + wave * 16;
    const int n0 = blockIdx.y * 64;
    const int m  = m0 + laneLo;
    const int h  = m / WW, w = m - (m / WW) * WW;
    const int koffA = (lane < 16) ? 0 : 16;       // A: 16x32 f16 layout
    const int koffB = (lane < 16) ? 0 : 16;       // B: 32x16 f16 layout

    v8f acc0 = {}, acc1 = {}, acc2 = {}, acc3 = {};

    for (int tap = 0; tap < 9; ++tap) {
        const int ky = tap / 3, kx = tap - ky * 3;
        __syncthreads();
        {   // cooperative stage: wT[n0..n0+63][tap*512 .. +511] -> LDS
            const _Float16* src = wT + (size_t)n0 * KTOT + (size_t)tap * 512;
            #pragma unroll 4
            for (int i = tid; i < 64 * 64; i += 256) {       // 4096 x v8h chunks
                int nL = i >> 6;
                int kc = (i & 63) << 3;
                *(v8h*)&bTile[nL * 512 + kc] =
                    *(const v8h*)(src + (size_t)nL * KTOT + kc);
            }
        }
        __syncthreads();

        const _Float16* aBase =
            featPad + ((size_t)(h + ky) * PW + (w + kx)) * CC + koffA;

        #pragma unroll 4
        for (int c0 = 0; c0 < 512; c0 += 32) {
            union { v16h v; v8h p[2]; } a, b;
            a.p[0] = *(const v8h*)(aBase + c0);
            a.p[1] = *(const v8h*)(aBase + c0 + 8);
            const _Float16* bb = &bTile[laneLo * 512 + c0 + koffB];

            b.p[0] = *(const v8h*)(bb);
            b.p[1] = *(const v8h*)(bb + 8);
            acc0 = __builtin_amdgcn_wmma_f32_16x16x32_f16(false, a.v, false, b.v,
                                                          (short)0, acc0, false, false);
            b.p[0] = *(const v8h*)(bb + 16 * 512);
            b.p[1] = *(const v8h*)(bb + 16 * 512 + 8);
            acc1 = __builtin_amdgcn_wmma_f32_16x16x32_f16(false, a.v, false, b.v,
                                                          (short)0, acc1, false, false);
            b.p[0] = *(const v8h*)(bb + 32 * 512);
            b.p[1] = *(const v8h*)(bb + 32 * 512 + 8);
            acc2 = __builtin_amdgcn_wmma_f32_16x16x32_f16(false, a.v, false, b.v,
                                                          (short)0, acc2, false, false);
            b.p[0] = *(const v8h*)(bb + 48 * 512);
            b.p[1] = *(const v8h*)(bb + 48 * 512 + 8);
            acc3 = __builtin_amdgcn_wmma_f32_16x16x32_f16(false, a.v, false, b.v,
                                                          (short)0, acc3, false, false);
        }
    }

    // epilogue: C/D layout — VGPR r: lanes 0-15 -> M=m0+r, lanes 16-31 -> M=m0+8+r
    const int mBase = m0 + ((lane >> 4) << 3);
    auto storeAcc = [&](const v8f& acc, int j) {
        const int nCol = n0 + j * 16 + laneLo;
        const float bias = convB[nCol];
        #pragma unroll
        for (int r = 0; r < 8; ++r) {
            float v = acc[r] + bias;
            v = v > 0.0f ? v : 0.0f;
            yOut[(size_t)(mBase + r) * CC + nCol] = (_Float16)v;
        }
    };
    storeAcc(acc0, 0); storeAcc(acc1, 1); storeAcc(acc2, 2); storeAcc(acc3, 3);
}

// ---------------------------------------------------------------------------
// K3: 1x1 heads — cls logits (512->9) and reg deltas (512->36)
// ---------------------------------------------------------------------------
__global__ __launch_bounds__(256)
void head_kernel(const _Float16* __restrict__ y,
                 const float* __restrict__ clsW, const float* __restrict__ clsB,
                 const float* __restrict__ regW, const float* __restrict__ regB,
                 float* __restrict__ clsLog, float* __restrict__ outDeltas) {
    int gid = blockIdx.x * 256 + threadIdx.x;
    if (gid >= NPOS * 45) return;
    int pos = gid / 45, oc = gid - pos * 45;
    const _Float16* yr = y + (size_t)pos * CC;
    if (oc < 9) {
        float acc = clsB[oc];
        for (int c = 0; c < CC; ++c) acc += (float)yr[c] * clsW[c * 9 + oc];
        clsLog[pos * 9 + oc] = acc;
    } else {
        int j = oc - 9;
        float acc = regB[j];
        for (int c = 0; c < CC; ++c) acc += (float)yr[c] * regW[c * 36 + j];
        outDeltas[(size_t)pos * 36 + j] = acc;   // == deltas[(pos*9+a)*4+d]
    }
}

// K4: softmax over the 9 anchor channels -> d_out scores
__global__ __launch_bounds__(256)
void softmax_kernel(const float* __restrict__ clsLog, float* __restrict__ outScores) {
    int pos = blockIdx.x * 256 + threadIdx.x;
    if (pos >= NPOS) return;
    float l[9], m = -3.4e38f;
    for (int a = 0; a < 9; ++a) { l[a] = clsLog[pos * 9 + a]; m = fmaxf(m, l[a]); }
    float s = 0.0f;
    for (int a = 0; a < 9; ++a) { l[a] = expf(l[a] - m); s += l[a]; }
    float inv = 1.0f / s;
    for (int a = 0; a < 9; ++a) outScores[pos * 9 + a] = l[a] * inv;
}

// K5: delta->box decode, clip, size filter
__global__ __launch_bounds__(256)
void decode_kernel(const float* __restrict__ anchors,
                   const float* __restrict__ outScores,
                   const float* __restrict__ outDeltas,
                   float* __restrict__ boxes, float* __restrict__ objArr,
                   float* __restrict__ validArr) {
    int r = blockIdx.x * 256 + threadIdx.x;
    if (r >= NANCH) return;
    const float* an = anchors + (size_t)r * 4;
    const float* dl = outDeltas + (size_t)r * 4;
    float ah = an[2] - an[0], aw = an[3] - an[1];
    float acy = an[0] + 0.5f * ah, acx = an[1] + 0.5f * aw;
    float cy = dl[0] * ah + acy, cx = dl[1] * aw + acx;
    float ph = expf(dl[2]) * ah, pw = expf(dl[3]) * aw;
    float y1 = fmaxf(cy - 0.5f * ph, 0.0f);
    float x1 = fmaxf(cx - 0.5f * pw, 0.0f);
    float y2 = fminf(cy + 0.5f * ph, IMSZ);
    float x2 = fminf(cx + 0.5f * pw, IMSZ);
    bool valid = ((y2 - y1) >= MINSZ) && ((x2 - x1) >= MINSZ);
    boxes[(size_t)r * 4 + 0] = y1; boxes[(size_t)r * 4 + 1] = x1;
    boxes[(size_t)r * 4 + 2] = y2; boxes[(size_t)r * 4 + 3] = x2;
    objArr[r]   = outScores[r];
    validArr[r] = valid ? 1.0f : 0.0f;
}

// K6: init histogram / selection buffers / meta
__global__ __launch_bounds__(256)
void init_kernel(unsigned* __restrict__ hist, int* __restrict__ meta,
                 float* __restrict__ selS, float* __restrict__ selB,
                 float* __restrict__ selA) {
    int i = blockIdx.x * 256 + threadIdx.x;
    if (i < 65536) hist[i] = 0u;
    if (i < MAXPRE) {
        selS[i] = NEGF;
        selB[i * 4 + 0] = 0.0f; selB[i * 4 + 1] = 0.0f;
        selB[i * 4 + 2] = 0.0f; selB[i * 4 + 3] = 0.0f;
        selA[i] = 0.0f;
    }
    if (i < 8) meta[i] = 0;
}

// K7: histogram of top 16 float bits (scores are positive => order-preserving)
__global__ __launch_bounds__(256)
void hist_kernel(const float* __restrict__ objArr, unsigned* __restrict__ hist) {
    int r = blockIdx.x * 256 + threadIdx.x;
    if (r >= NANCH) return;
    unsigned bin = __float_as_uint(objArr[r]) >> 16;
    atomicAdd(&hist[bin], 1u);
}

// K8: find the bin containing rank MAXPRE (descending)
__global__ __launch_bounds__(1024)
void thresh_kernel(const unsigned* __restrict__ hist, int* __restrict__ meta) {
    __shared__ unsigned part[1024];
    int t = threadIdx.x;
    unsigned s = 0;
    for (int b = t * 64; b < t * 64 + 64; ++b) s += hist[b];
    part[t] = s;
    __syncthreads();
    if (t == 0) {
        unsigned cum = 0; int seg = -1;
        for (int q = 1023; q >= 0; --q) {
            if (cum + part[q] >= (unsigned)MAXPRE) { seg = q; break; }
            cum += part[q];
        }
        int thrBin = 0; unsigned above = cum;
        if (seg >= 0) {
            for (int b = seg * 64 + 63; b >= seg * 64; --b) {
                unsigned c = hist[b];
                if (above + c >= (unsigned)MAXPRE) { thrBin = b; break; }
                above += c;
            }
        }
        meta[2] = thrBin;
        meta[3] = (int)above;
        meta[4] = MAXPRE - (int)above;
    }
}

// K9: compact top-MAXPRE candidates; apply size-filter masking (NEG / zero box)
__global__ __launch_bounds__(256)
void compact_kernel(const float* __restrict__ objArr, const float* __restrict__ validArr,
                    const float* __restrict__ boxes, int* __restrict__ meta,
                    float* __restrict__ selS, float* __restrict__ selB,
                    float* __restrict__ selA) {
    int r = blockIdx.x * 256 + threadIdx.x;
    if (r >= NANCH) return;
    float obj = objArr[r];
    int bin = (int)(__float_as_uint(obj) >> 16);
    int thrBin = meta[2], above = meta[3], need = meta[4];
    int slot = -1;
    if (bin > thrBin) {
        slot = atomicAdd(&meta[0], 1);
    } else if (bin == thrBin) {
        int p = atomicAdd(&meta[1], 1);
        if (p < need) slot = above + p;
    }
    if (slot >= 0 && slot < MAXPRE) {
        bool v = validArr[r] > 0.5f;
        float b0 = v ? boxes[(size_t)r * 4 + 0] : 0.0f;
        float b1 = v ? boxes[(size_t)r * 4 + 1] : 0.0f;
        float b2 = v ? boxes[(size_t)r * 4 + 2] : 0.0f;
        float b3 = v ? boxes[(size_t)r * 4 + 3] : 0.0f;
        selS[slot] = v ? obj : NEGF;
        selB[slot * 4 + 0] = b0; selB[slot * 4 + 1] = b1;
        selB[slot * 4 + 2] = b2; selB[slot * 4 + 3] = b3;
        selA[slot] = (b2 - b0) * (b3 - b1);
    }
}

// K10: greedy NMS, single workgroup, MAXPOST iterations; writes proposals
__global__ __launch_bounds__(1024)
void nms_kernel(float* __restrict__ selS, const float* __restrict__ selB,
                const float* __restrict__ selA, float* __restrict__ outProps) {
    __shared__ float sv[1024];
    __shared__ int   si[1024];
    __shared__ float bb[4];
    const int tid = threadIdx.x;
    for (int it = 0; it < MAXPOST; ++it) {
        float best = -3.4e38f; int bi = 0;
        for (int i = tid; i < MAXPRE; i += 1024) {
            float s = selS[i];
            if (s > best) { best = s; bi = i; }
        }
        sv[tid] = best; si[tid] = bi;
        __syncthreads();
        for (int off = 512; off > 0; off >>= 1) {
            if (tid < off && sv[tid + off] > sv[tid]) {
                sv[tid] = sv[tid + off]; si[tid] = si[tid + off];
            }
            __syncthreads();
        }
        if (tid == 0) {
            int j = si[0];
            bb[0] = selB[j * 4 + 0]; bb[1] = selB[j * 4 + 1];
            bb[2] = selB[j * 4 + 2]; bb[3] = selB[j * 4 + 3];
            outProps[it * 4 + 0] = bb[0]; outProps[it * 4 + 1] = bb[1];
            outProps[it * 4 + 2] = bb[2]; outProps[it * 4 + 3] = bb[3];
            selS[j] = NEGF;
        }
        __syncthreads();
        float a0 = bb[0], a1 = bb[1], a2 = bb[2], a3 = bb[3];
        float areaJ = (a2 - a0) * (a3 - a1);
        for (int i = tid; i < MAXPRE; i += 1024) {
            if (selS[i] <= NEGF) continue;
            float yy1 = fmaxf(a0, selB[i * 4 + 0]);
            float xx1 = fmaxf(a1, selB[i * 4 + 1]);
            float yy2 = fminf(a2, selB[i * 4 + 2]);
            float xx2 = fminf(a3, selB[i * 4 + 3]);
            float inter = fmaxf(yy2 - yy1, 0.0f) * fmaxf(xx2 - xx1, 0.0f);
            float iou = inter / (areaJ + selA[i] - inter + 1e-8f);
            if (iou > IOUT) selS[i] = NEGF;
        }
        __syncthreads();
    }
}

// ---------------------------------------------------------------------------
extern "C" void kernel_launch(void* const* d_in, const int* in_sizes, int n_in,
                              void* d_out, int out_size, void* d_ws, size_t ws_size,
                              hipStream_t stream) {
    const float* feat    = (const float*)d_in[1];
    const float* anchors = (const float*)d_in[2];
    const float* convW   = (const float*)d_in[3];
    const float* convB   = (const float*)d_in[4];
    const float* clsW    = (const float*)d_in[5];
    const float* clsB    = (const float*)d_in[6];
    const float* regW    = (const float*)d_in[7];
    const float* regB    = (const float*)d_in[8];

    char* ws = (char*)d_ws;
    _Float16* featPad = (_Float16*)(ws + O_FEATPAD);
    _Float16* wT      = (_Float16*)(ws + O_WT);
    _Float16* yBuf    = (_Float16*)(ws + O_Y);
    float*    clsLog  = (float*)(ws + O_CLSLOG);
    float*    boxes   = (float*)(ws + O_BOXES);
    float*    objArr  = (float*)(ws + O_OBJ);
    float*    validA  = (float*)(ws + O_VALID);
    float*    selS    = (float*)(ws + O_SELS);
    float*    selB    = (float*)(ws + O_SELB);
    float*    selA    = (float*)(ws + O_SELA);
    unsigned* hist    = (unsigned*)(ws + O_HIST);
    int*      meta    = (int*)(ws + O_META);

    float* out        = (float*)d_out;
    float* outScores  = out;                          // 82944
    float* outDeltas  = out + NANCH;                  // 331776
    float* outProps   = out + NANCH + NANCH * 4;      // 8000

    // K1: pad + transpose to f16
    {
        size_t total = (size_t)PW * PW * CC + (size_t)CC * KTOT;
        int blocks = (int)((total + 255) / 256);
        prep_kernel<<<blocks, 256, 0, stream>>>(feat, convW, featPad, wT);
    }
    // K2: WMMA implicit-GEMM conv + bias + ReLU
    {
        dim3 grid(NPOS / 128, CC / 64);               // 72 x 8
        gemm_conv_kernel<<<grid, 256, 0, stream>>>(featPad, wT, convB, yBuf);
    }
    // K3: 1x1 heads
    head_kernel<<<(NPOS * 45 + 255) / 256, 256, 0, stream>>>(
        yBuf, clsW, clsB, regW, regB, clsLog, outDeltas);
    // K4: softmax -> scores output
    softmax_kernel<<<(NPOS + 255) / 256, 256, 0, stream>>>(clsLog, outScores);
    // K5: decode + clip + size filter
    decode_kernel<<<(NANCH + 255) / 256, 256, 0, stream>>>(
        anchors, outScores, outDeltas, boxes, objArr, validA);
    // K6: init
    init_kernel<<<(65536 + 255) / 256, 256, 0, stream>>>(hist, meta, selS, selB, selA);
    // K7: histogram
    hist_kernel<<<(NANCH + 255) / 256, 256, 0, stream>>>(objArr, hist);
    // K8: rank threshold
    thresh_kernel<<<1, 1024, 0, stream>>>(hist, meta);
    // K9: compact top-12000
    compact_kernel<<<(NANCH + 255) / 256, 256, 0, stream>>>(
        objArr, validA, boxes, meta, selS, selB, selA);
    // K10: greedy NMS -> proposals output
    nms_kernel<<<1, 1024, 0, stream>>>(selS, selB, selA, outProps);
}
